// RecursiveNN_8555574854002
// MI455X (gfx1250) — matrix operand order repaired and checked by reading, not compiled, not run on hardware
//
#include <hip/hip_runtime.h>
#include <cstdint>

#define N_LEAVES 131072
#define N_NODES  262143
#define EMBED    300
#define KDIM     600       // 2*EMBED
#define NPAD     304       // 19 * 16
#define KPAD     608       // 19 * 32
#define NT_TILES 19
#define KT_TILES 19

typedef __attribute__((ext_vector_type(16))) __bf16 v16bf;
typedef __attribute__((ext_vector_type(8)))  float  v8f;

union Frag { v16bf v; unsigned int u[8]; };

__device__ __forceinline__ float wave_reduce_add(float v) {
    v += __shfl_xor(v, 16, 32);
    v += __shfl_xor(v, 8, 32);
    v += __shfl_xor(v, 4, 32);
    v += __shfl_xor(v, 2, 32);
    v += __shfl_xor(v, 1, 32);
    return v;
}

// Pack W_w [300,600] f32 row-major into bf16 B-fragments.
// Element index t = ((nt*19 + kt)*32 + lane)*16 + i, where for tile (nt,kt):
// lane c in [0,16): N = nt*16+c, K = kt*32 + i ; lane c+16: same N, K = kt*32+16+i.
__global__ void prep_weights_kernel(const float* __restrict__ Ww,
                                    __bf16* __restrict__ Bpack) {
    int t = blockIdx.x * 256 + threadIdx.x;
    int i    = t & 15;
    int lane = (t >> 4) & 31;
    int kt   = (t >> 9) % KT_TILES;
    int nt   = (t >> 9) / KT_TILES;
    if (nt >= NT_TILES) return;
    int e = nt * 16 + (lane & 15);
    int k = kt * 32 + ((lane & 16) ? 16 : 0) + i;
    float v = (e < EMBED && k < KDIM) ? Ww[(long long)e * KDIM + k] : 0.0f;
    Bpack[t] = (__bf16)v;
}

// One wave per leaf: gather emb row, L2-normalize, emit bf16 level-0,
// fused logits + weighted CE contribution.
__global__ void leaf_kernel(const int* __restrict__ leaf_ids,
                            const float* __restrict__ emb,
                            __bf16* __restrict__ lvl0,
                            const float* __restrict__ Pw,
                            const float* __restrict__ Pb,
                            const int* __restrict__ labels,
                            const int* __restrict__ annot,
                            float* __restrict__ logits,
                            float* __restrict__ loss) {
    __shared__ float lossAcc;
    if (threadIdx.x == 0) lossAcc = 0.0f;
    __syncthreads();
    int wave = threadIdx.x >> 5, lane = threadIdx.x & 31;
    int node = blockIdx.x * 8 + wave;
    if (node < N_LEAVES) {
        const float* row = emb + (long long)leaf_ids[node] * EMBED;
        float s = 0.0f;
        for (int e = lane; e < EMBED; e += 32) { float v = row[e]; s += v * v; }
        s = wave_reduce_add(s);
        float rn = rsqrtf(s);
        float p0 = 0.0f, p1 = 0.0f;
        for (int e = lane; e < EMBED; e += 32) {
            float h = row[e] * rn;
            lvl0[(long long)node * EMBED + e] = (__bf16)h;
            p0 += Pw[e] * h;
            p1 += Pw[EMBED + e] * h;
        }
        p0 = wave_reduce_add(p0);
        p1 = wave_reduce_add(p1);
        if (lane == 0) {
            float l0 = p0 + Pb[0], l1 = p1 + Pb[1];
            logits[2 * node]     = l0;
            logits[2 * node + 1] = l1;
            float mx  = fmaxf(l0, l1);
            float lse = mx + logf(expf(l0 - mx) + expf(l1 - mx));
            float lp  = (labels[node] == 0 ? l0 : l1) - lse;
            float w   = (annot[node] == 1) ? 0.7f : 0.3f;
            atomicAdd(&lossAcc, -w * lp);
        }
    }
    __syncthreads();
    if (threadIdx.x == 0) atomicAdd(loss, lossAcc);
}

// One block = 16 output nodes. Stage x = concat(cur[2m], cur[2m+1]) (zero-padded
// to 608) in LDS. kt-outer / nt-inner: each wave loads ONE A fragment per K-step
// from LDS and feeds it to the WMMAs of its 2-3 owned N-tiles. All B fragments
// for the step are issued as one load clause BEFORE the WMMAs so the waits can
// drain while earlier WMMAs execute. Then tanh+bias tiles land in LDS fp32;
// cross-feature L2 norm, bf16 store of next level, fused logits + CE.
__global__ void level_kernel(const __bf16* __restrict__ cur,
                             __bf16* __restrict__ next,
                             const __bf16* __restrict__ Bpack,
                             const float* __restrict__ bias,
                             const float* __restrict__ Pw,
                             const float* __restrict__ Pb,
                             const int* __restrict__ labels,
                             const int* __restrict__ annot,
                             float* __restrict__ logits,
                             float* __restrict__ loss,
                             int n_out, int off_out) {
    __shared__ __align__(16) __bf16 xs[16][KPAD];   // 19456 B
    __shared__ __align__(16) float  ys[16][NPAD];   // 19456 B
    __shared__ float lossAcc;

    int tid = threadIdx.x;
    if (tid == 0) lossAcc = 0.0f;
    int m0 = blockIdx.x * 16;

    // Fill xs with 32-bit (bf16 pair) coalesced loads; rows/ks out of range -> 0.
    unsigned int* xs32 = (unsigned int*)&xs[0][0];
    for (int t = tid; t < 16 * (KPAD / 2); t += 256) {
        int m  = t / (KPAD / 2);
        int j  = t % (KPAD / 2);
        int k0 = 2 * j;
        int mo = m0 + m;
        unsigned int v = 0u;
        if (mo < n_out && k0 < KDIM) {
            long long src = (k0 < EMBED)
                ? ((long long)(2 * mo) * EMBED + k0)
                : ((long long)(2 * mo + 1) * EMBED + (k0 - EMBED));
            v = *(const unsigned int*)(cur + src);
        }
        xs32[t] = v;
    }
    __syncthreads();

    int wave = tid >> 5, lane = tid & 31;
    int row   = lane & 15;
    int kqoff = (lane & 16) ? 8 : 0;           // A-fragment K sub-offset (in dwords)
    int mbase = (lane < 16) ? 0 : 8;           // C/D tile row base per lane half

    // N-tile ownership: every wave owns nt0 = w and nt1 = w+8; waves 0..2 also
    // own nt2 = w+16 (8*2 + 3 = 19 tiles). All guards are wave-uniform, so EXEC
    // is all-ones around every WMMA.
    const int nt0 = wave;
    const int nt1 = wave + 8;
    const int nt2 = wave + 16;
    const bool has2 = (nt2 < NT_TILES);

    v8f acc0 = {0.f, 0.f, 0.f, 0.f, 0.f, 0.f, 0.f, 0.f};
    v8f acc1 = {0.f, 0.f, 0.f, 0.f, 0.f, 0.f, 0.f, 0.f};
    v8f acc2 = {0.f, 0.f, 0.f, 0.f, 0.f, 0.f, 0.f, 0.f};

    // Per-lane fragment base for nt0; nt stride of 8 tiles = 8*19*512 bf16.
    const unsigned int* bq0 =
        (const unsigned int*)(Bpack + ((long long)nt0 * KT_TILES * 32 + lane) * 16);
    const int ntStrideQ = 8 * KT_TILES * 512 / 2;   // dwords between nt and nt+8

#pragma unroll 1
    for (int kt = 0; kt < KT_TILES; ++kt) {
        // One A fragment per K-step, shared by all owned N-tiles.
        Frag a;
        int abase = row * (KPAD / 2) + kt * 16 + kqoff;
#pragma unroll
        for (int j = 0; j < 4; ++j) {
            a.u[j]     = xs32[abase + j];        // K = kb..kb+7
            a.u[j + 4] = xs32[abase + 4 + j];    // K = kb+8..kb+15
        }
        const unsigned int* bq = bq0 + kt * 256; // 512 bf16 per (nt,kt) tile
        // Unconditional prefetch of next K-slab (kt==18 lands in valid ws memory).
        __builtin_prefetch(bq + 256, 0, 1);

        // Issue ALL B-fragment loads for this K-step up front (single clause),
        // then run the WMMAs; later waits overlap with earlier WMMA execution.
        Frag b0, b1, b2;
#pragma unroll
        for (int j = 0; j < 8; ++j) b0.u[j] = bq[j];
#pragma unroll
        for (int j = 0; j < 8; ++j) b1.u[j] = bq[ntStrideQ + j];
        if (has2) {
#pragma unroll
            for (int j = 0; j < 8; ++j) b2.u[j] = bq[2 * ntStrideQ + j];
        }

        acc0 = __builtin_amdgcn_wmma_f32_16x16x32_bf16(
            false, a.v, false, b0.v, (short)0, acc0, false, false);
        acc1 = __builtin_amdgcn_wmma_f32_16x16x32_bf16(
            false, a.v, false, b1.v, (short)0, acc1, false, false);
        if (has2) {
            acc2 = __builtin_amdgcn_wmma_f32_16x16x32_bf16(
                false, a.v, false, b2.v, (short)0, acc2, false, false);
        }
    }

    // Epilogue: bias + tanh into fp32 LDS tile (padded cols 300..303 -> 0).
    auto emit_tile = [&](const v8f& acc, int nt) {
        int ecol = nt * 16 + (lane & 15);
#pragma unroll
        for (int r = 0; r < 8; ++r) {
            float v = 0.0f;
            if (ecol < EMBED) v = tanhf(acc[r] + bias[ecol]);
            ys[mbase + r][ecol] = v;
        }
    };
    emit_tile(acc0, nt0);
    emit_tile(acc1, nt1);
    if (has2) emit_tile(acc2, nt2);
    __syncthreads();

    // Norm + logits: wave w handles rows w and w+8.
    for (int rr = 0; rr < 2; ++rr) {
        int row2 = wave + 8 * rr;
        int mo   = m0 + row2;
        if (mo >= n_out) continue;
        float s = 0.0f;
        for (int e = lane; e < NPAD; e += 32) { float v = ys[row2][e]; s += v * v; }
        s = wave_reduce_add(s);
        float rn = rsqrtf(s);
        float p0 = 0.0f, p1 = 0.0f;
        for (int e = lane; e < EMBED; e += 32) {
            float h = ys[row2][e] * rn;
            next[(long long)mo * EMBED + e] = (__bf16)h;
            p0 += Pw[e] * h;
            p1 += Pw[EMBED + e] * h;
        }
        p0 = wave_reduce_add(p0);
        p1 = wave_reduce_add(p1);
        if (lane == 0) {
            int gnode = off_out + mo;
            float l0 = p0 + Pb[0], l1 = p1 + Pb[1];
            logits[2 * gnode]     = l0;
            logits[2 * gnode + 1] = l1;
            float mx  = fmaxf(l0, l1);
            float lse = mx + logf(expf(l0 - mx) + expf(l1 - mx));
            float lp  = (labels[gnode] == 0 ? l0 : l1) - lse;
            float w   = (annot[gnode] == 1) ? 0.7f : 0.3f;
            atomicAdd(&lossAcc, -w * lp);
        }
    }
    __syncthreads();
    if (tid == 0) atomicAdd(loss, lossAcc);
}

extern "C" void kernel_launch(void* const* d_in, const int* in_sizes, int n_in,
                              void* d_out, int out_size, void* d_ws, size_t ws_size,
                              hipStream_t stream) {
    (void)in_sizes; (void)n_in; (void)out_size; (void)ws_size;
    const int*   leaf_ids = (const int*)d_in[0];
    const int*   labels   = (const int*)d_in[1];
    const int*   annot    = (const int*)d_in[2];
    const float* emb      = (const float*)d_in[3];
    const float* Ww       = (const float*)d_in[4];
    const float* Wb       = (const float*)d_in[5];
    const float* Pw       = (const float*)d_in[6];
    const float* Pb       = (const float*)d_in[7];

    float* logits = (float*)d_out;
    float* loss   = logits + 2LL * N_NODES;   // tuple: logits flat, then scalar loss

    // Workspace layout (all offsets 256B-aligned):
    //   Bpack: 19*19*512 bf16 = 369,664 B
    //   lvlA : 131072*300 bf16 = 78,643,200 B
    //   lvlB :  65536*300 bf16 = 39,321,600 B
    char* ws = (char*)d_ws;
    __bf16* Bpack = (__bf16*)ws;
    __bf16* lvlA  = (__bf16*)(ws + 369664);
    __bf16* lvlB  = (__bf16*)(ws + 369664 + 78643200LL);

    hipMemsetAsync(loss, 0, sizeof(float), stream);

    // 19*19*512 = 184,832 elements = 722 blocks of 256.
    prep_weights_kernel<<<722, 256, 0, stream>>>(Ww, Bpack);

    leaf_kernel<<<N_LEAVES / 8, 256, 0, stream>>>(
        leaf_ids, emb, lvlA, Pw, Pb, labels, annot, logits, loss);

    __bf16* bufs[2] = {lvlA, lvlB};
    int cb = 0, n = N_LEAVES, off = 0;
    for (int lvl = 0; lvl < 17; ++lvl) {
        int n_out   = n >> 1;
        int off_out = off + n;
        level_kernel<<<(n_out + 15) / 16, 256, 0, stream>>>(
            bufs[cb], bufs[cb ^ 1], Bpack, Wb, Pw, Pb,
            labels, annot, logits, loss, n_out, off_out);
        n = n_out; off = off_out; cb ^= 1;
    }
}
